// EGNNScore_26809185861851
// MI455X (gfx1250) — matrix operand order repaired and checked by reading, not compiled, not run on hardware
//
#include <hip/hip_runtime.h>

// ---------------------------------------------------------------------------
// EGNN score model for MI455X (gfx1250, wave32, WMMA).
// All GEMMs run through v_wmma_f32_16x16x32_bf16 with f32 accumulation.
// Activations/weights staged as bf16 (halves HBM traffic vs f32 reference);
// reductions, coords and residual streams stay f32.
// GEMM waves compute 32x64 output tiles: two A fragments share four B
// fragments -> 8 WMMAs per 12 b128 loads per K-step.
// ---------------------------------------------------------------------------

typedef __bf16 bhalf;
typedef bhalf bvec16 __attribute__((ext_vector_type(16)));
typedef bhalf bvec8  __attribute__((ext_vector_type(8)));
typedef float fvec8  __attribute__((ext_vector_type(8)));

constexpr int Dh = 256;    // node feature dim (== H)
constexpr int De = 64;     // edge attr dim
constexpr int Nn = 512;    // nodes per sample
constexpr int KE = 608;    // 2*Dh + 1 + De = 577, padded to multiple of 32
constexpr int KN = 512;    // node concat dim (Dh + Dh)
constexpr float CSCALE = 0.1f;

__device__ __forceinline__ float silu_f(float v) {
  return v / (1.0f + __expf(-v));
}

// ------------------------------- init --------------------------------------
__global__ void k_init(const float* __restrict__ h, const float* __restrict__ x,
                       float* __restrict__ bb, float* __restrict__ coord, int NT) {
  int stride = gridDim.x * blockDim.x;
  for (int i = blockIdx.x * blockDim.x + threadIdx.x; i < NT * Dh; i += stride)
    bb[i] = h[i];
  for (int i = blockIdx.x * blockDim.x + threadIdx.x; i < NT * 3; i += stride)
    coord[i] = x[i] * CSCALE;
}

// ---- weight convert: W[K,Nc] f32 row-major -> Wt[Nc,Kp] bf16 (zero pad) ----
__global__ void k_wconv(const float* __restrict__ W, bhalf* __restrict__ Wt,
                        int K, int Nc, int Kp) {
  int stride = gridDim.x * blockDim.x;
  int total = Nc * Kp;
  for (int i = blockIdx.x * blockDim.x + threadIdx.x; i < total; i += stride) {
    int n = i / Kp, k = i % Kp;
    float v = (k < K) ? W[(size_t)k * Nc + n] : 0.0f;
    Wt[i] = (bhalf)v;
  }
}

// ------------------------- per-row edge counts -----------------------------
__global__ void k_count(const int* __restrict__ row, float* __restrict__ cnt, int E) {
  int stride = gridDim.x * blockDim.x;
  for (int e = blockIdx.x * blockDim.x + threadIdx.x; e < E; e += stride)
    atomicAdd(&cnt[row[e]], 1.0f);
}

// -------------- edge feature gather: one wave per edge ---------------------
// feat[e] = [ h[row] (256) | h[col] (256) | radial (1) | e_attr (64) | 0 pad ]
__global__ void k_edge_feat(const int* __restrict__ row, const int* __restrict__ col,
                            const float* __restrict__ bb, const float* __restrict__ coord,
                            const float* __restrict__ eattr,
                            bhalf* __restrict__ feat, float* __restrict__ cdiff, int E) {
  int lane = threadIdx.x & 31;
  int wid  = (blockIdx.x * blockDim.x + threadIdx.x) >> 5;
  int nw   = (gridDim.x * blockDim.x) >> 5;
  for (int e = wid; e < E; e += nw) {
    int r = row[e], c = col[e];
    float dx = coord[r * 3 + 0] - coord[c * 3 + 0];
    float dy = coord[r * 3 + 1] - coord[c * 3 + 1];
    float dz = coord[r * 3 + 2] - coord[c * 3 + 2];
    float rad = dx * dx + dy * dy + dz * dz;
    if (lane == 0) {
      cdiff[e * 4 + 0] = dx; cdiff[e * 4 + 1] = dy;
      cdiff[e * 4 + 2] = dz; cdiff[e * 4 + 3] = rad;
    }
    bhalf* f = feat + (size_t)e * KE;
    const float* hr = bb + (size_t)r * Dh;
    const float* hc = bb + (size_t)c * Dh;
    for (int j = lane; j < Dh; j += 32) {
      f[j]      = (bhalf)hr[j];
      f[Dh + j] = (bhalf)hc[j];
    }
    if (lane == 0) f[2 * Dh] = (bhalf)rad;
    const float* ea =
        eattr + (((size_t)(r / Nn) * Nn + (size_t)(r % Nn)) * Nn + (size_t)(c % Nn)) * De;
    for (int j = lane; j < De; j += 32) f[2 * Dh + 1 + j] = (bhalf)ea[j];
    for (int j = 2 * Dh + 1 + De + lane; j < KE; j += 32) f[j] = (bhalf)0.0f;
  }
}

// ------------------------------- WMMA GEMM ---------------------------------
// C[M,Nc] = act(A[M,Kp] @ Bt[Nc,Kp]^T + bias). One wave per 32x64 tile:
// two A fragments (rows m0, m0+16) share four B fragments along N, so each
// K-step performs 8 WMMAs on 12 b128 loads.
// Fragment layouts used here (wave32; contiguous k, same for A and B):
//   A 16x32: lane m=L%16, half=L>>4; elem j -> k = half*16+j
//   B 32x16: lane n=L%16, half=L>>4; elem j -> k = half*16+j  (from Bt[n][k])
//   C 16x16: vgpr r -> row = half*8+r, col = n
template <bool SILU, typename OutT>
__global__ __launch_bounds__(256) void
k_gemm(const bhalf* __restrict__ A, const bhalf* __restrict__ Bt,
       const float* __restrict__ bias, OutT* __restrict__ C,
       int M, int Nc, int Kp) {
  const int lane = threadIdx.x & 31;
  const int n16  = lane & 15;
  const int half = lane >> 4;
  const int wid  = (blockIdx.x * blockDim.x + threadIdx.x) >> 5;
  const int nw   = (gridDim.x * blockDim.x) >> 5;
  const int mt = M >> 5;          // 32-row tiles
  const int ng = Nc >> 6;         // 64-col groups
  const int tasks = mt * ng;
  for (int task = wid; task < tasks; task += nw) {
    const int m0 = (task / ng) << 5;
    const int n0 = (task % ng) << 6;
    fvec8 acc0[4] = {};
    fvec8 acc1[4] = {};
    const bhalf* arow0 = A + (size_t)(m0 + n16) * Kp + half * 16;
    const bhalf* arow1 = arow0 + (size_t)16 * Kp;
    const bhalf* brow  = Bt + (size_t)(n0 + n16) * Kp + half * 16;
    for (int k0 = 0; k0 < Kp; k0 += 32) {
      bvec16 af0, af1;
      {
        bvec8 lo = *(const bvec8*)(arow0 + k0);
        bvec8 hi = *(const bvec8*)(arow0 + k0 + 8);
#pragma unroll
        for (int j = 0; j < 8; ++j) { af0[j] = lo[j]; af0[j + 8] = hi[j]; }
      }
      {
        bvec8 lo = *(const bvec8*)(arow1 + k0);
        bvec8 hi = *(const bvec8*)(arow1 + k0 + 8);
#pragma unroll
        for (int j = 0; j < 8; ++j) { af1[j] = lo[j]; af1[j + 8] = hi[j]; }
      }
      if (k0 + 32 < Kp) {
        __builtin_prefetch(arow0 + k0 + 32, 0, 1);
        __builtin_prefetch(arow1 + k0 + 32, 0, 1);
      }
#pragma unroll
      for (int t = 0; t < 4; ++t) {
        bvec16 bfr = *(const bvec16*)(brow + (size_t)t * 16 * Kp + k0);
        acc0[t] = __builtin_amdgcn_wmma_f32_16x16x32_bf16(
            false, af0, false, bfr, (short)0, acc0[t], false, false);
        acc1[t] = __builtin_amdgcn_wmma_f32_16x16x32_bf16(
            false, af1, false, bfr, (short)0, acc1[t], false, false);
      }
    }
#pragma unroll
    for (int t = 0; t < 4; ++t) {
      int colc = n0 + t * 16 + n16;
      float bv = bias[colc];
#pragma unroll
      for (int r = 0; r < 8; ++r) {
        float v0 = acc0[t][r] + bv;
        float v1 = acc1[t][r] + bv;
        if (SILU) { v0 = silu_f(v0); v1 = silu_f(v1); }
        C[(size_t)(m0 + half * 8 + r) * Nc + colc]      = (OutT)v0;
        C[(size_t)(m0 + 16 + half * 8 + r) * Nc + colc] = (OutT)v1;
      }
    }
  }
}

// -------- edge scatter: t = c1 . wc2 ; num += cdiff*t ; msum += m ----------
__global__ void k_edge_scatter(const int* __restrict__ row, const bhalf* __restrict__ m,
                               const bhalf* __restrict__ c1, const float* __restrict__ wc2,
                               const float* __restrict__ cdiff,
                               float* __restrict__ msum, float* __restrict__ num, int E) {
  int lane = threadIdx.x & 31;
  int wid  = (blockIdx.x * blockDim.x + threadIdx.x) >> 5;
  int nw   = (gridDim.x * blockDim.x) >> 5;
  for (int e = wid; e < E; e += nw) {
    const bhalf* ce = c1 + (size_t)e * Dh;
    float part = 0.0f;
    for (int j = lane; j < Dh; j += 32) part += (float)ce[j] * wc2[j];
#pragma unroll
    for (int off = 16; off > 0; off >>= 1) part += __shfl_xor(part, off, 32);
    int r = row[e];
    if (lane < 3) atomicAdd(&num[r * 4 + lane], cdiff[e * 4 + lane] * part);
    const bhalf* me = m + (size_t)e * Dh;
    float* ms = msum + (size_t)r * Dh;
    for (int j = lane; j < Dh; j += 32) atomicAdd(&ms[j], (float)me[j]);
  }
}

// coord = scale * (coord + num / max(cnt,1))   (scale folds the aliasing 2x/3x)
__global__ void k_coord_update(float* __restrict__ coord, const float* __restrict__ num,
                               const float* __restrict__ cnt, float scale, int NT) {
  int stride = gridDim.x * blockDim.x;
  for (int i = blockIdx.x * blockDim.x + threadIdx.x; i < NT * 3; i += stride) {
    int node = i / 3, k = i % 3;
    float c = fmaxf(cnt[node], 1.0f);
    coord[i] = scale * (coord[i] + num[node * 4 + k] / c);
  }
}

// nodecat = [ bb | msum ] as bf16
__global__ void k_concat(const float* __restrict__ bb, const float* __restrict__ msum,
                         bhalf* __restrict__ out, int NT) {
  int stride = gridDim.x * blockDim.x;
  for (int i = blockIdx.x * blockDim.x + threadIdx.x; i < NT * KN; i += stride) {
    int node = i / KN, j = i % KN;
    float v = (j < Dh) ? bb[(size_t)node * Dh + j] : msum[(size_t)node * Dh + (j - Dh)];
    out[i] = (bhalf)v;
  }
}

// bb = fac*bb + outA (+ outB)
__global__ void k_combine(float* __restrict__ bb, const float* __restrict__ oA,
                          const float* __restrict__ oB, float fac, int n) {
  int stride = gridDim.x * blockDim.x;
  for (int i = blockIdx.x * blockDim.x + threadIdx.x; i < n; i += stride) {
    float v = fac * bb[i] + oA[i];
    if (oB) v += oB[i];
    bb[i] = v;
  }
}

// mean-pool 16 nodes -> [B*NBB, 256], then @ final_W[256,3] + final_b
__global__ __launch_bounds__(256) void
k_final(const float* __restrict__ bb, const float* __restrict__ fW,
        const float* __restrict__ fb, float* __restrict__ out) {
  __shared__ float pooled[Dh];
  int g = blockIdx.x;        // b*32 + backbone group
  int d = threadIdx.x;       // 0..255
  const float* base = bb + (size_t)g * 16 * Dh;
  float s = 0.0f;
#pragma unroll
  for (int r = 0; r < 16; ++r) s += base[r * Dh + d];
  pooled[d] = s * (1.0f / 16.0f);
  __syncthreads();
  if (d < 3) {
    float acc = fb[d];
    for (int k = 0; k < Dh; ++k) acc += pooled[k] * fW[k * 3 + d];
    out[g * 3 + d] = acc;
  }
}

// ---------------------------------------------------------------------------
extern "C" void kernel_launch(void* const* d_in, const int* in_sizes, int n_in,
                              void* d_out, int out_size, void* d_ws, size_t ws_size,
                              hipStream_t stream) {
  (void)n_in; (void)out_size; (void)ws_size;
  const float* h    = (const float*)d_in[0];
  const float* x    = (const float*)d_in[1];
  const float* ea   = (const float*)d_in[2];
  const float* We1  = (const float*)d_in[3];
  const float* be1  = (const float*)d_in[4];
  const float* We2  = (const float*)d_in[5];
  const float* be2  = (const float*)d_in[6];
  const float* Wn1  = (const float*)d_in[7];
  const float* bn1  = (const float*)d_in[8];
  const float* Wn2  = (const float*)d_in[9];
  const float* bn2  = (const float*)d_in[10];
  const float* Wc1  = (const float*)d_in[11];
  const float* bc1  = (const float*)d_in[12];
  const float* Wc2  = (const float*)d_in[13];
  const float* finW = (const float*)d_in[14];
  const float* finB = (const float*)d_in[15];
  const int*  intra = (const int*)d_in[16];
  const int*  inter = (const int*)d_in[17];

  const int E  = in_sizes[16] / 2;   // 65536
  const int NT = in_sizes[0] / Dh;   // 2048

  // -------- workspace carve-up (~166 MB; fits in the 192 MB global L2) -----
  char* ws = (char*)d_ws;
  size_t off = 0;
  auto carve = [&](size_t bytes) -> char* {
    char* p = ws + off;
    off += (bytes + 255) & ~(size_t)255;
    return p;
  };
  float* bb    = (float*)carve((size_t)NT * Dh * 4);
  float* coord = (float*)carve((size_t)NT * 3 * 4);
  float* outA  = (float*)carve((size_t)NT * Dh * 4);
  float* outB  = (float*)carve((size_t)NT * Dh * 4);
  float* msum  = (float*)carve((size_t)NT * Dh * 4);
  float* num   = (float*)carve((size_t)NT * 4 * 4);
  float* cntI  = (float*)carve((size_t)NT * 4);
  float* cntE  = (float*)carve((size_t)NT * 4);
  float* cdiff = (float*)carve((size_t)E * 4 * 4);
  bhalf* feat  = (bhalf*)carve((size_t)E * KE * 2);
  bhalf* m1    = (bhalf*)carve((size_t)E * Dh * 2);   // reused as c1
  bhalf* mm    = (bhalf*)carve((size_t)E * Dh * 2);
  bhalf* ncat  = (bhalf*)carve((size_t)NT * KN * 2);
  bhalf* nh    = (bhalf*)carve((size_t)NT * Dh * 2);
  bhalf* We1t  = (bhalf*)carve((size_t)7 * Dh * KE * 2);
  bhalf* We2t  = (bhalf*)carve((size_t)7 * Dh * Dh * 2);
  bhalf* Wc1t  = (bhalf*)carve((size_t)7 * Dh * Dh * 2);
  bhalf* Wn1t  = (bhalf*)carve((size_t)7 * Dh * KN * 2);
  bhalf* Wn2t  = (bhalf*)carve((size_t)7 * Dh * Dh * 2);

  // -------- init node state + convert/transposed bf16 weights --------------
  k_init<<<2048, 256, 0, stream>>>(h, x, bb, coord, NT);
  for (int l = 0; l < 7; ++l) {
    k_wconv<<<(Dh * KE + 255) / 256, 256, 0, stream>>>(
        We1 + (size_t)l * 577 * Dh, We1t + (size_t)l * Dh * KE, 577, Dh, KE);
    k_wconv<<<(Dh * Dh + 255) / 256, 256, 0, stream>>>(
        We2 + (size_t)l * Dh * Dh, We2t + (size_t)l * Dh * Dh, Dh, Dh, Dh);
    k_wconv<<<(Dh * Dh + 255) / 256, 256, 0, stream>>>(
        Wc1 + (size_t)l * Dh * Dh, Wc1t + (size_t)l * Dh * Dh, Dh, Dh, Dh);
    k_wconv<<<(Dh * KN + 255) / 256, 256, 0, stream>>>(
        Wn1 + (size_t)l * KN * Dh, Wn1t + (size_t)l * Dh * KN, KN, Dh, KN);
    k_wconv<<<(Dh * Dh + 255) / 256, 256, 0, stream>>>(
        Wn2 + (size_t)l * Dh * Dh, Wn2t + (size_t)l * Dh * Dh, Dh, Dh, Dh);
  }

  // -------- edge counts (index-only; constant across layers) ---------------
  hipMemsetAsync(cntI, 0, (size_t)NT * 4, stream);
  hipMemsetAsync(cntE, 0, (size_t)NT * 4, stream);
  k_count<<<(E + 255) / 256, 256, 0, stream>>>(intra, cntI, E);
  k_count<<<(E + 255) / 256, 256, 0, stream>>>(inter, cntE, E);

  const int* introw = intra;      const int* intcol = intra + E;
  const int* interow = inter;     const int* intecol = inter + E;

  const int featBlocks = (E * 32 + 255) / 256;            // wave per edge
  const int eblocks = ((E / 32) * (Dh / 64) + 7) / 8;     // wave per 32x64 tile
  const int nblocks = ((NT / 32) * (Dh / 64) + 7) / 8;

  auto run_egcl = [&](int pl, const int* r, const int* c, const float* cnt,
                      float cscale, float* outbuf) {
    hipMemsetAsync(msum, 0, (size_t)NT * Dh * 4, stream);
    hipMemsetAsync(num, 0, (size_t)NT * 4 * 4, stream);
    k_edge_feat<<<featBlocks, 256, 0, stream>>>(r, c, bb, coord, ea, feat, cdiff, E);
    // edge MLP: m = silu(silu(feat@We1+be1)@We2+be2)
    k_gemm<true, bhalf><<<eblocks, 256, 0, stream>>>(
        feat, We1t + (size_t)pl * Dh * KE, be1 + pl * Dh, m1, E, Dh, KE);
    k_gemm<true, bhalf><<<eblocks, 256, 0, stream>>>(
        m1, We2t + (size_t)pl * Dh * Dh, be2 + pl * Dh, mm, E, Dh, Dh);
    // coord MLP hidden: c1 = silu(m@Wc1+bc1)   (m1 buffer reused)
    k_gemm<true, bhalf><<<eblocks, 256, 0, stream>>>(
        mm, Wc1t + (size_t)pl * Dh * Dh, bc1 + pl * Dh, m1, E, Dh, Dh);
    // t = c1 @ Wc2 ; scatter trans and messages
    k_edge_scatter<<<featBlocks, 256, 0, stream>>>(
        r, mm, m1, Wc2 + pl * Dh, cdiff, msum, num, E);
    k_coord_update<<<(NT * 3 + 255) / 256, 256, 0, stream>>>(coord, num, cnt, cscale, NT);
    // node MLP: out = silu([bb|msum]@Wn1+bn1)@Wn2+bn2
    k_concat<<<(NT * KN + 255) / 256, 256, 0, stream>>>(bb, msum, ncat, NT);
    k_gemm<true, bhalf><<<nblocks, 256, 0, stream>>>(
        ncat, Wn1t + (size_t)pl * Dh * KN, bn1 + pl * Dh, nh, NT, Dh, KN);
    k_gemm<false, float><<<nblocks, 256, 0, stream>>>(
        nh, Wn2t + (size_t)pl * Dh * Dh, bn2 + pl * Dh, outbuf, NT, Dh, Dh);
  };

  for (int i = 0; i < 4; ++i) {
    if (i < 3) {
      run_egcl(i, introw, intcol, cntI, 1.0f, outA);          // coord -> c1
      run_egcl(4 + i, interow, intecol, cntE, 3.0f, outB);    // coord -> 3*c2
      // bb + (bb+outA) + (bb+outB) = 3*bb + outA + outB
      k_combine<<<2048, 256, 0, stream>>>(bb, outA, outB, 3.0f, NT * Dh);
    } else {
      run_egcl(i, introw, intcol, cntI, 2.0f, outA);          // coord -> 2*c1
      k_combine<<<2048, 256, 0, stream>>>(bb, outA, nullptr, 2.0f, NT * Dh);
    }
  }

  k_final<<<NT / 16, 256, 0, stream>>>(bb, finW, finB, (float*)d_out);
}
